// CrossAttentionDown_41996190221023
// MI455X (gfx1250) — matrix operand order repaired and checked by reading, not compiled, not run on hardware
//
#include <hip/hip_runtime.h>

typedef _Float16 v16h __attribute__((ext_vector_type(16)));
typedef _Float16 v8h  __attribute__((ext_vector_type(8)));
typedef _Float16 v4h  __attribute__((ext_vector_type(4)));
typedef float    v8f  __attribute__((ext_vector_type(8)));

// Problem dims (from setup_inputs in the reference)
constexpr int B_ = 4, H_ = 16, T_ = 4096, D_ = 64, MAXN_ = 32, P_ = 128;
constexpr int NS = 4;             // T-segments (flash-decoding split for occupancy)
constexpr int TSEG = T_ / NS;     // 1024
constexpr int TC = 64;            // t-chunk per iteration
constexpr int KP = 72;            // LDS row pitch in halfs (144B: 16B-aligned, conflict-free)
constexpr int PSTR = 72;          // partial row stride in floats: [0..63]=O', [64]=m(log2), [65]=l
constexpr int SSTR = P_ * PSTR;   // partial segment stride
constexpr float LOG2E = 1.4426950408889634f;
constexpr float QK_SCALE2 = 0.125f * LOG2E;       // (1/sqrt(64)) * log2(e): softmax in 2^x domain
constexpr float L2T_16 = 0.83048202372184059f;    // log2(10000)/16
constexpr float NEG_BIG = -1e30f;                 // finite "-inf"

__device__ __forceinline__ v16h combine16(v8h lo, v8h hi) {
  v16h r;
#pragma unroll
  for (int i = 0; i < 8; ++i) { r[i] = lo[i]; r[i + 8] = hi[i]; }
  return r;
}

// Butterfly max over the 16-lane group via DPP16 (VALU only; pairings stay in-half):
// quad_perm(1,0,3,2), quad_perm(2,3,0,1), row_half_mirror, row_mirror
__device__ __forceinline__ float dpp_max16(float x) {
  int xi = __builtin_bit_cast(int, x);
  x = fmaxf(x, __builtin_bit_cast(float, __builtin_amdgcn_update_dpp(0, xi, 0xB1, 0xF, 0xF, true)));
  xi = __builtin_bit_cast(int, x);
  x = fmaxf(x, __builtin_bit_cast(float, __builtin_amdgcn_update_dpp(0, xi, 0x4E, 0xF, 0xF, true)));
  xi = __builtin_bit_cast(int, x);
  x = fmaxf(x, __builtin_bit_cast(float, __builtin_amdgcn_update_dpp(0, xi, 0x141, 0xF, 0xF, true)));
  xi = __builtin_bit_cast(int, x);
  x = fmaxf(x, __builtin_bit_cast(float, __builtin_amdgcn_update_dpp(0, xi, 0x140, 0xF, 0xF, true)));
  return x;
}

// ---------------- Kernel 1: region starts[b][n] = first t with regions[b,t]==n+1, else 0
__global__ void ca_region_starts(const int* __restrict__ regions, int* __restrict__ starts) {
  __shared__ int smin[MAXN_ + 1];
  const int b = blockIdx.x, tid = threadIdx.x;
  if (tid <= MAXN_) smin[tid] = T_;
  __syncthreads();
  for (int t = tid; t < T_; t += blockDim.x) {
    int r = regions[b * T_ + t];
    if (r >= 1 && r <= MAXN_) atomicMin(&smin[r], t);
  }
  __syncthreads();
  if (tid < MAXN_) {
    int s = smin[tid + 1];
    starts[b * MAXN_ + tid] = (s == T_) ? 0 : s;
  }
}

// ---------------- Kernel 2: region-RoPE q -> f16 workspace (row-major b,h,p,d)
__global__ void ca_qrope(const float* __restrict__ q, const int* __restrict__ starts,
                         _Float16* __restrict__ qh) {
  int idx = blockIdx.x * blockDim.x + threadIdx.x;   // one thread per (b,h,p, d-pair)
  if (idx >= B_ * H_ * P_ * 32) return;
  const int j2 = idx & 31;
  const int p  = (idx >> 5) & (P_ - 1);
  const int bh = idx >> 12;           // 32*128 = 2^12
  const int b  = bh >> 4;             // H = 16
  float pos; int j;
  if (j2 < 16) { pos = (float)starts[b * MAXN_ + (p >> 2)]; j = j2; }       // gpos half
  else         { pos = (float)((p >> 2) + 1);               j = j2 - 16; }  // ridx half
  const float inv = exp2f(-(float)j * L2T_16);
  float sn, cs; __sincosf(pos * inv, &sn, &cs);
  const float* src = q + ((size_t)bh * P_ + p) * D_ + 2 * j2;
  const float x = src[0], y = src[1];
  _Float16* dst = qh + ((size_t)bh * P_ + p) * D_ + 2 * j2;
  dst[0] = (_Float16)(x * cs - y * sn);
  dst[1] = (_Float16)(x * sn + y * cs);
}

// ---------------- Kernel 3: flash attention over one T-segment.
// One block per (b,h,seg); 8 waves, wave w owns P rows [16w,16w+16). Writes partial (O', m, l).
__launch_bounds__(256)
__global__ void ca_attn(const float* __restrict__ xk, const float* __restrict__ xv,
                        const int* __restrict__ regions, const unsigned char* __restrict__ tmask,
                        const unsigned char* __restrict__ nmask,
                        const float* __restrict__ bias_same, const float* __restrict__ bias_diff,
                        const _Float16* __restrict__ qh, float* __restrict__ part) {
  __shared__ __align__(16) _Float16 k_sm[TC * KP];          // roped keys, [t][d]
  __shared__ __align__(16) _Float16 v_sm[(D_ + 16) * KP];   // values^T [d][t]; rows 64..79 = ones column trick
  __shared__ __align__(16) _Float16 p_sm[8][16 * KP];       // per-wave prob tile, [p_local][t]
  __shared__ int reg_sm[TC];
  __shared__ unsigned char tm_sm[TC];

  const int bh  = blockIdx.x >> 2;
  const int seg = blockIdx.x & (NS - 1);
  const int b = bh >> 4, h = bh & 15;
  const int tid = threadIdx.x;
  const int wid = tid >> 5;
  const int lane = tid & 31;
  const int half = lane >> 4;     // 16-lane group
  const int n = lane & 15;        // column / A-row index within tile

  const float bs2 = bias_same[h] * LOG2E, bd2 = bias_diff[h] * LOG2E;  // log2-domain bias

  // Static "ones-column" tail of V^T (d=64 -> 1.0, d=65..79 -> 0): PV-WMMA tile 4 computes rowsum(P~)
  for (int idx = tid; idx < 16 * TC; idx += 256) {
    const int r = idx >> 6, c = idx & (TC - 1);
    v_sm[(D_ + r) * KP + c] = (r == 0) ? (_Float16)1.f : (_Float16)0.f;
  }

  // Q A-fragments (16x32 f16, two k-steps over D=64). A row m = lane&15.
  v16h a0, a1;
  {
    const _Float16* qbase = qh + ((size_t)bh * P_ + wid * 16 + n) * D_;
    v8h q0 = *(const v8h*)(qbase + half * 8);
    v8h q1 = *(const v8h*)(qbase + 16 + half * 8);
    v8h q2 = *(const v8h*)(qbase + 32 + half * 8);
    v8h q3 = *(const v8h*)(qbase + 48 + half * 8);
    a0 = combine16(q0, q1);
    a1 = combine16(q2, q3);
  }

  // Per-row (C/D row = v + half*8) constants and softmax state (m in log2 domain)
  int ridxrow[8]; bool nmr[8];
  float mrow[8];
  v8f oacc[5] = {v8f{}, v8f{}, v8f{}, v8f{}, v8f{}};   // [4] accumulates rowsum(P~) in col n==0
#pragma unroll
  for (int v = 0; v < 8; ++v) {
    const int prow = wid * 16 + v + half * 8;
    const int nidx = prow >> 2;
    ridxrow[v] = nidx + 1;
    nmr[v] = nmask[b * MAXN_ + nidx] != 0;
    mrow[v] = NEG_BIG;
  }

  const size_t kvbase = (size_t)bh * T_ * D_;
  const int tbeg = seg * TSEG, tend = tbeg + TSEG;

  for (int t0 = tbeg; t0 < tend; t0 += TC) {
    __syncthreads();   // previous chunk fully consumed (also covers ones-tail init)

    // -------- cooperative stage: rope K -> f16 LDS [t][d]; V -> f16 LDS [d][t]
    if (tid < TC) {
      reg_sm[tid] = regions[b * T_ + t0 + tid];
      tm_sm[tid]  = tmask[b * T_ + t0 + tid];
    }
#pragma unroll
    for (int i = 0; i < 4; ++i) {
      const int qid = i * 256 + tid;          // 1024 float4 per chunk (2 RoPE pairs each)
      const int t = qid >> 4;
      const int d = (qid & 15) * 4;           // d%4==0 -> both pairs in same RoPE half
      const float4 kx = *(const float4*)(xk + kvbase + (size_t)(t0 + t) * D_ + d);
      float pos; int jb;
      if (d < 32) { pos = (float)(t0 + t); jb = d >> 1; }                    // gpos half
      else { pos = (float)regions[b * T_ + t0 + t]; jb = (d - 32) >> 1; }    // region half
      const float inv0 = exp2f(-(float)jb * L2T_16);
      const float inv1 = exp2f(-(float)(jb + 1) * L2T_16);
      float s0, c0, s1, c1;
      __sincosf(pos * inv0, &s0, &c0);
      __sincosf(pos * inv1, &s1, &c1);
      v4h kk;
      kk[0] = (_Float16)(kx.x * c0 - kx.y * s0);
      kk[1] = (_Float16)(kx.x * s0 + kx.y * c0);
      kk[2] = (_Float16)(kx.z * c1 - kx.w * s1);
      kk[3] = (_Float16)(kx.z * s1 + kx.w * c1);
      *(v4h*)(k_sm + t * KP + d) = kk;        // one ds_store_b64
      const float4 vx = *(const float4*)(xv + kvbase + (size_t)(t0 + t) * D_ + d);
      v_sm[(d + 0) * KP + t] = (_Float16)vx.x;   // transpose -> b16 stores
      v_sm[(d + 1) * KP + t] = (_Float16)vx.y;
      v_sm[(d + 2) * KP + t] = (_Float16)vx.z;
      v_sm[(d + 3) * KP + t] = (_Float16)vx.w;
    }
    if (t0 + TC < tend) {  // pull next chunk toward L2 (global_prefetch_b8)
      __builtin_prefetch(xk + kvbase + (size_t)(t0 + TC) * D_ + tid * 16, 0, 0);
      __builtin_prefetch(xv + kvbase + (size_t)(t0 + TC) * D_ + tid * 16, 0, 0);
    }
    __syncthreads();

    // -------- scores S' = (q.k^T/8 + bias) * log2e  for this wave's 16 rows x 64 cols
    float sv[4][8];
#pragma unroll
    for (int sub = 0; sub < 4; ++sub) {
      const int tl = sub * 16 + n;                  // this lane's key column
      const _Float16* kr = k_sm + tl * KP;
      v16h b0 = combine16(*(const v8h*)(kr + half * 8),
                          *(const v8h*)(kr + 16 + half * 8));
      v16h b1 = combine16(*(const v8h*)(kr + 32 + half * 8),
                          *(const v8h*)(kr + 48 + half * 8));
      v8f acc = {};
      acc = __builtin_amdgcn_wmma_f32_16x16x32_f16(false, a0, false, b0, (short)0, acc, false, false);
      acc = __builtin_amdgcn_wmma_f32_16x16x32_f16(false, a1, false, b1, (short)0, acc, false, false);
      const int rt = reg_sm[tl];
      const bool tm = tm_sm[tl] != 0;
#pragma unroll
      for (int v = 0; v < 8; ++v) {
        float s = fmaf(acc[v], QK_SCALE2, (ridxrow[v] == rt) ? bs2 : bd2);
        if (!tm || !nmr[v]) s = NEG_BIG;
        sv[sub][v] = s;
      }
    }

    // -------- online softmax in 2^x domain: row max via DPP butterfly (no LDS traffic)
    float mnew[8], scl[8];
#pragma unroll
    for (int v = 0; v < 8; ++v) {
      float rm = fmaxf(fmaxf(sv[0][v], sv[1][v]), fmaxf(sv[2][v], sv[3][v]));
      rm = dpp_max16(rm);
      mnew[v] = fmaxf(mrow[v], rm);
      scl[v] = exp2f(mrow[v] - mnew[v]);
      mrow[v] = mnew[v];
#pragma unroll
      for (int nt = 0; nt < 5; ++nt) oacc[nt][v] *= scl[v];
    }
#pragma unroll
    for (int sub = 0; sub < 4; ++sub)
#pragma unroll
      for (int v = 0; v < 8; ++v) {
        const float p = exp2f(sv[sub][v] - mnew[v]);
        p_sm[wid][(v + half * 8) * KP + sub * 16 + n] = (_Float16)p;   // C-layout -> row major
      }

    // cross-lane LDS transpose hazard: drain DS before re-reading our own tile
    asm volatile("s_wait_dscnt 0" ::: "memory");

    // -------- O += P~ . [V | ones]   (tile nt=4 accumulates rowsum(P~) -> l, in col n==0)
    const _Float16* pw = p_sm[wid];
#pragma unroll
    for (int kt = 0; kt < 2; ++kt) {
      const _Float16* pr = pw + n * KP + kt * 32;   // A row m = lane&15
      v16h pa = combine16(*(const v8h*)(pr + half * 8),
                          *(const v8h*)(pr + 16 + half * 8));
#pragma unroll
      for (int nt = 0; nt < 5; ++nt) {
        const _Float16* vr = v_sm + (nt * 16 + n) * KP + kt * 32;
        v16h vb = combine16(*(const v8h*)(vr + half * 8),
                            *(const v8h*)(vr + 16 + half * 8));
        oacc[nt] = __builtin_amdgcn_wmma_f32_16x16x32_f16(false, pa, false, vb, (short)0, oacc[nt], false, false);
      }
    }
  }

  // -------- write unnormalized partial (O', m_log2, l) for this segment
  float* pb = part + (size_t)(bh * NS + seg) * SSTR;
#pragma unroll
  for (int v = 0; v < 8; ++v) {
    const int prow = wid * 16 + v + half * 8;
    float* o = pb + prow * PSTR + n;
#pragma unroll
    for (int nt = 0; nt < 4; ++nt) o[nt * 16] = oacc[nt][v];
    if (n == 0) {
      pb[prow * PSTR + 64] = mrow[v];      // log2-domain running max
      pb[prow * PSTR + 65] = oacc[4][v];   // l = rowsum of probs (col 0 of ones tile)
    }
  }
}

// ---------------- Kernel 4: merge NS segment partials (m is log2-domain): out = sum w_s O'_s / sum w_s l_s
__global__ void ca_combine(const float* __restrict__ part, float* __restrict__ out) {
  const int row = blockIdx.x;            // bh*P + prow
  const int d = threadIdx.x;             // 0..63
  const int bh = row >> 7, prow = row & (P_ - 1);
  const float* pr = part + (size_t)bh * NS * SSTR + prow * PSTR;
  float M = NEG_BIG;
#pragma unroll
  for (int s = 0; s < NS; ++s) M = fmaxf(M, pr[s * SSTR + 64]);
  float L = 0.f, acc = 0.f;
#pragma unroll
  for (int s = 0; s < NS; ++s) {
    const float w = exp2f(pr[s * SSTR + 64] - M);
    L += w * pr[s * SSTR + 65];
    acc += w * pr[s * SSTR + d];
  }
  out[(size_t)row * D_ + d] = (L > 0.f) ? acc / L : 0.f;
}

extern "C" void kernel_launch(void* const* d_in, const int* in_sizes, int n_in,
                              void* d_out, int out_size, void* d_ws, size_t ws_size,
                              hipStream_t stream) {
  (void)in_sizes; (void)n_in; (void)out_size; (void)ws_size;
  const float* q   = (const float*)d_in[0];
  const float* xk  = (const float*)d_in[1];
  const float* xv  = (const float*)d_in[2];
  const int*   reg = (const int*)d_in[3];
  const unsigned char* tm = (const unsigned char*)d_in[4];
  const unsigned char* nm = (const unsigned char*)d_in[5];
  // d_in[6] = max_n scalar (constant 32 here)
  const float* bsm = (const float*)d_in[7];
  const float* bdf = (const float*)d_in[8];
  float* out = (float*)d_out;

  int*      starts = (int*)d_ws;                            // B*32 ints
  _Float16* qh     = (_Float16*)((char*)d_ws + 1024);       // B*H*P*D halfs (1 MB)
  float*    part   = (float*)((char*)d_ws + 1024 + (size_t)B_ * H_ * P_ * D_ * 2);  // ~9.4 MB

  ca_region_starts<<<B_, 256, 0, stream>>>(reg, starts);
  ca_qrope<<<(B_ * H_ * P_ * 32 + 255) / 256, 256, 0, stream>>>(q, starts, qh);
  ca_attn<<<B_ * H_ * NS, 256, 0, stream>>>(xk, xv, reg, tm, nm, bsm, bdf, qh, part);
  ca_combine<<<B_ * H_ * P_, 64, 0, stream>>>(part, out);
}
